// LocalSelfAttention_36747740184719
// MI455X (gfx1250) — matrix operand order repaired and checked by reading, not compiled, not run on hardware
//
#include <hip/hip_runtime.h>
#include <math.h>

// ---------------------------------------------------------------------------
// LocalSelfAttention on gfx1250 (MI455X): LN -> QKV GEMM (bf16 WMMA) ->
// windowed attention (W=3) -> out-proj GEMM (bf16 WMMA, f32 out + bias).
// Data staging uses CDNA5 async global->LDS copies (ASYNCcnt) with
// double-buffered K tiles in the GEMMs.
// ---------------------------------------------------------------------------

typedef __bf16 bf16;
typedef __attribute__((ext_vector_type(8)))  __bf16 bf16x8;
typedef __attribute__((ext_vector_type(16))) __bf16 bf16x16;
typedef __attribute__((ext_vector_type(8)))  float  floatx8;

#define BATCH 4
#define SEQ   4096
#define DIM   768
#define HEADS 12
#define HDIM  64
#define WIN   3
#define ROWS  (BATCH * SEQ)   // 16384

// Flip to 0 to fall back to load+ds_store staging if the assembler rejects
// the async mnemonics.
#ifndef USE_ASYNC_LDS
#define USE_ASYNC_LDS 1
#endif

// 16-byte global -> LDS copy. Async version: per-lane
// GLOBAL_LOAD_ASYNC_TO_LDS_B128 (GV mode), tracked with ASYNCcnt.
// VDST VGPR carries the LDS byte offset = low 32 bits of the generic pointer.
__device__ inline void copy16_to_lds(const bf16* g, bf16* l) {
#if USE_ASYNC_LDS
  asm volatile("global_load_async_to_lds_b128 %0, %1, off"
               :
               : "v"((unsigned int)(unsigned long long)l),
                 "v"((unsigned long long)g)
               : "memory");
#else
  *(uint4*)l = *(const uint4*)g;
#endif
}

__device__ inline void wait_async_copies() {
#if USE_ASYNC_LDS
  asm volatile("s_wait_asynccnt 0x0" ::: "memory");
#endif
}

// ---------------------------------------------------------------------------
// LayerNorm: one block per row (768 cols), write bf16 normalized row.
// ---------------------------------------------------------------------------
__global__ __launch_bounds__(256) void ln_kernel(
    const float* __restrict__ x, const float* __restrict__ g,
    const float* __restrict__ b, bf16* __restrict__ xn) {
  __shared__ float red[256];
  const int tid = threadIdx.x;
  const float* xr = x + (size_t)blockIdx.x * DIM;

  float v0 = xr[tid], v1 = xr[tid + 256], v2 = xr[tid + 512];
  red[tid] = v0 + v1 + v2;
  __syncthreads();
  #pragma unroll
  for (int o = 128; o > 0; o >>= 1) {
    if (tid < o) red[tid] += red[tid + o];
    __syncthreads();
  }
  const float mu = red[0] * (1.0f / DIM);
  __syncthreads();

  const float d0 = v0 - mu, d1 = v1 - mu, d2 = v2 - mu;
  red[tid] = d0 * d0 + d1 * d1 + d2 * d2;
  __syncthreads();
  #pragma unroll
  for (int o = 128; o > 0; o >>= 1) {
    if (tid < o) red[tid] += red[tid + o];
    __syncthreads();
  }
  const float rstd = rsqrtf(red[0] * (1.0f / DIM) + 1e-5f);

  bf16* o = xn + (size_t)blockIdx.x * DIM;
  o[tid]       = (bf16)(d0 * rstd * g[tid]       + b[tid]);
  o[tid + 256] = (bf16)(d1 * rstd * g[tid + 256] + b[tid + 256]);
  o[tid + 512] = (bf16)(d2 * rstd * g[tid + 512] + b[tid + 512]);
}

// ---------------------------------------------------------------------------
// Transpose f32 (R x C) -> bf16 (C x R), 32x32 LDS tiles.
// ---------------------------------------------------------------------------
__global__ __launch_bounds__(256) void transpose_to_bf16(
    const float* __restrict__ in, bf16* __restrict__ out, int R, int C) {
  __shared__ float t[32][33];
  const int tx = threadIdx.x & 31;
  const int ty = threadIdx.x >> 5;   // 0..7
  const int c0 = blockIdx.x * 32;
  const int r0 = blockIdx.y * 32;
  #pragma unroll
  for (int i = 0; i < 32; i += 8)
    t[ty + i][tx] = in[(size_t)(r0 + ty + i) * C + (c0 + tx)];
  __syncthreads();
  #pragma unroll
  for (int i = 0; i < 32; i += 8)
    out[(size_t)(c0 + ty + i) * R + (r0 + tx)] = (bf16)t[tx][ty + i];
}

// ---------------------------------------------------------------------------
// WMMA fragment helper: concat two 16B LDS chunks into a 16-elem bf16 frag.
// ---------------------------------------------------------------------------
__device__ inline bf16x16 make_frag(const bf16* lo, const bf16* hi) {
  bf16x8 a = *(const bf16x8*)lo;
  bf16x8 b = *(const bf16x8*)hi;
  bf16x16 r;
  #pragma unroll
  for (int e = 0; e < 8; ++e) { r[e] = a[e]; r[e + 8] = b[e]; }
  return r;
}

// ---------------------------------------------------------------------------
// bf16 GEMM, C = A(MxK) * Bt(NxK)^T, K = 768. 128x128 block tile, 8 waves
// (wave32) in a 2x4 grid, each wave computes 64x32 via 4x2 WMMA 16x16x32.
// K tiles are double-buffered in LDS and staged with async copies.
// MODE 0: scatter D (bf16) into deinterleaved q/k/v buffers (QKV epilogue).
// MODE 1: D (f32) + bias -> outF.
// ---------------------------------------------------------------------------
template <int MODE>
__global__ __launch_bounds__(256) void gemm_bf16_kernel(
    const bf16* __restrict__ A, const bf16* __restrict__ Bt,
    int Ncols,
    float* __restrict__ outF, const float* __restrict__ bias,
    bf16* __restrict__ qb, bf16* __restrict__ kb, bf16* __restrict__ vb) {
  constexpr int K   = 768;
  constexpr int TK  = 32;
  constexpr int NT  = K / TK;  // 24 K-steps
  constexpr int LDW = 40;      // bf16 row stride in LDS: 80B, 16B-aligned

  __shared__ __attribute__((aligned(16))) bf16 As[2][128 * LDW];
  __shared__ __attribute__((aligned(16))) bf16 Bs[2][128 * LDW];

  const int tid  = threadIdx.x;
  const int lane = tid & 31;
  const int wid  = tid >> 5;
  const int wm   = wid & 1;        // 0..1  -> 64-row slab
  const int wn   = wid >> 1;       // 0..3  -> 32-col slab
  const int lm   = lane & 15;
  const int m0   = blockIdx.y * 128;
  const int n0   = blockIdx.x * 128;

  floatx8 acc[4][2];
  #pragma unroll
  for (int mi = 0; mi < 4; ++mi)
    #pragma unroll
    for (int ni = 0; ni < 2; ++ni)
      #pragma unroll
      for (int e = 0; e < 8; ++e) acc[mi][ni][e] = 0.0f;

  const int halfA = (lane < 16) ? 0 : 8;    // A: K 0-7/16-23 vs 8-15/24-31
  const int halfB = (lane < 16) ? 0 : 8;    // B: K 0-7/16-23 vs 8-15/24-31

  // Stage one 128x32 A tile + 128x32 B tile: 512 x 16B chunks each,
  // 4 chunks per thread.
  auto fill_tile = [&](int kt, int buf) {
    #pragma unroll
    for (int it = 0; it < 2; ++it) {
      const int c   = tid + it * 256;
      const int row = c >> 2;
      const int j   = c & 3;
      copy16_to_lds(A + (size_t)(m0 + row) * K + kt + j * 8,
                    &As[buf][row * LDW + j * 8]);
      copy16_to_lds(Bt + (size_t)(n0 + row) * K + kt + j * 8,
                    &Bs[buf][row * LDW + j * 8]);
    }
  };

  fill_tile(0, 0);

  for (int t = 0; t < NT; ++t) {
    wait_async_copies();       // our tile-t copies done
    __syncthreads();           // everyone's tile-t copies visible
    const int buf = t & 1;

    if (t + 1 < NT) fill_tile((t + 1) * TK, buf ^ 1);
    if (t + 2 < NT) {          // warm L2 two tiles ahead (global_prefetch_b8)
      __builtin_prefetch(A + (size_t)(m0 + (tid >> 1)) * K + (t + 2) * TK, 0, 1);
      __builtin_prefetch(Bt + (size_t)(n0 + (tid >> 1)) * K + (t + 2) * TK, 0, 1);
    }

    bf16x16 af[4], bfr[2];
    #pragma unroll
    for (int mi = 0; mi < 4; ++mi) {
      const bf16* p = &As[buf][(wm * 64 + mi * 16 + lm) * LDW + halfA];
      af[mi] = make_frag(p, p + 16);
    }
    #pragma unroll
    for (int ni = 0; ni < 2; ++ni) {
      const bf16* p = &Bs[buf][(wn * 32 + ni * 16 + lm) * LDW + halfB];
      bfr[ni] = make_frag(p, p + 16);
    }

    #pragma unroll
    for (int mi = 0; mi < 4; ++mi)
      #pragma unroll
      for (int ni = 0; ni < 2; ++ni)
        acc[mi][ni] = __builtin_amdgcn_wmma_f32_16x16x32_bf16(
            false, af[mi], false, bfr[ni], (short)0, acc[mi][ni], false, false);
    // Tile t+2's copies are only issued after the next barrier, which no wave
    // passes until its WMMAs consumed tile t: no second barrier needed.
  }

  // Epilogue. D layout: lanes 0-15 -> M = base+0..7, lanes 16-31 -> M = base+8..15.
  const int rhalf = (lane >> 4) * 8;
  #pragma unroll
  for (int mi = 0; mi < 4; ++mi) {
    #pragma unroll
    for (int ni = 0; ni < 2; ++ni) {
      const int col  = n0 + wn * 32 + ni * 16 + lm;
      const int rowb = m0 + wm * 64 + mi * 16 + rhalf;
      if constexpr (MODE == 1) {
        const float bb = bias[col];
        #pragma unroll
        for (int r = 0; r < 8; ++r)
          outF[(size_t)(rowb + r) * Ncols + col] = acc[mi][ni][r] + bb;
      } else {
        // qkv column c -> (h, d, s): c = h*192 + d*3 + s
        const int h = col / 192;
        const int rem = col - h * 192;
        const int d = rem / 3;
        const int s = rem - d * 3;
        bf16* dst = (s == 0) ? qb : ((s == 1) ? kb : vb);
        const size_t off = (size_t)h * HDIM + d;
        #pragma unroll
        for (int r = 0; r < 8; ++r)
          dst[(size_t)(rowb + r) * DIM + off] = (bf16)acc[mi][ni][r];
      }
    }
  }
}

// ---------------------------------------------------------------------------
// Windowed attention: block = 128 threads = 128 query positions of one (b,h).
// q/k/v windows staged into LDS with async copies; per-thread softmax over 7.
// Output written as bf16 in (b, n, H*D) layout, ready for out-proj GEMM.
// ---------------------------------------------------------------------------
__global__ __launch_bounds__(128) void attn_kernel(
    const bf16* __restrict__ q, const bf16* __restrict__ k,
    const bf16* __restrict__ v, const float* __restrict__ log_temp,
    bf16* __restrict__ out) {
  constexpr int LDA = 72;  // bf16 row stride: 144B, 16B-aligned
  __shared__ __attribute__((aligned(16))) bf16 Qs[128 * LDA];
  __shared__ __attribute__((aligned(16))) bf16 Ks[134 * LDA];
  __shared__ __attribute__((aligned(16))) bf16 Vs[134 * LDA];

  const int tid = threadIdx.x;
  const int bh  = blockIdx.y;
  const int b   = bh / HEADS;
  const int h   = bh - b * HEADS;
  const int n0  = blockIdx.x * 128;
  const size_t base = (size_t)b * SEQ * DIM + (size_t)h * HDIM;

  {
    const bf16* src = q + base + (size_t)(n0 + tid) * DIM;
    #pragma unroll
    for (int j = 0; j < 8; ++j)
      copy16_to_lds(src + j * 8, &Qs[tid * LDA + j * 8]);
  }
  for (int r = tid; r < 134; r += 128) {
    int g = n0 - WIN + r;
    g = g < 0 ? 0 : (g > SEQ - 1 ? SEQ - 1 : g);
    const bf16* ks = k + base + (size_t)g * DIM;
    const bf16* vs = v + base + (size_t)g * DIM;
    #pragma unroll
    for (int j = 0; j < 8; ++j) {
      copy16_to_lds(ks + j * 8, &Ks[r * LDA + j * 8]);
      copy16_to_lds(vs + j * 8, &Vs[r * LDA + j * 8]);
    }
  }
  wait_async_copies();
  __syncthreads();

  const float temp = expf(log_temp[0]);
  const int n = n0 + tid;

  float dots[7];
  float mx = -3.0e38f;
  #pragma unroll
  for (int w = 0; w < 7; ++w) {
    const int idx = n + w - WIN;
    float s = 0.0f;
    #pragma unroll
    for (int j = 0; j < 8; ++j) {
      bf16x8 qv = *(const bf16x8*)(&Qs[tid * LDA + j * 8]);
      bf16x8 kv = *(const bf16x8*)(&Ks[(tid + w) * LDA + j * 8]);
      #pragma unroll
      for (int e = 0; e < 8; ++e) s += (float)qv[e] * (float)kv[e];
    }
    const bool ok = (idx >= 0) && (idx < SEQ);
    dots[w] = ok ? s * temp : -3.0e38f;
    mx = fmaxf(mx, dots[w]);
  }

  float p[7];
  float sum = 0.0f;
  #pragma unroll
  for (int w = 0; w < 7; ++w) { p[w] = expf(dots[w] - mx); sum += p[w]; }
  const float inv = 1.0f / sum;
  #pragma unroll
  for (int w = 0; w < 7; ++w) p[w] *= inv;

  bf16* dst = out + base + (size_t)n * DIM;
  #pragma unroll
  for (int j = 0; j < 8; ++j) {
    float a[8] = {0, 0, 0, 0, 0, 0, 0, 0};
    #pragma unroll
    for (int w = 0; w < 7; ++w) {
      bf16x8 vv = *(const bf16x8*)(&Vs[(tid + w) * LDA + j * 8]);
      #pragma unroll
      for (int e = 0; e < 8; ++e) a[e] += p[w] * (float)vv[e];
    }
    #pragma unroll
    for (int e = 0; e < 8; ++e) dst[j * 8 + e] = (bf16)a[e];
  }
}

// ---------------------------------------------------------------------------
// Launcher
// ---------------------------------------------------------------------------
extern "C" void kernel_launch(void* const* d_in, const int* in_sizes, int n_in,
                              void* d_out, int out_size, void* d_ws, size_t ws_size,
                              hipStream_t stream) {
  const float* x        = (const float*)d_in[0];
  const float* ln_scale = (const float*)d_in[1];
  const float* ln_bias  = (const float*)d_in[2];
  const float* w_qkv    = (const float*)d_in[3];
  const float* w_out    = (const float*)d_in[4];
  const float* b_out    = (const float*)d_in[5];
  const float* log_temp = (const float*)d_in[6];
  float* out = (float*)d_out;

  // Workspace layout (bf16), ~130.5 MB total.
  bf16* xn    = (bf16*)d_ws;
  bf16* wqkvT = xn    + (size_t)ROWS * DIM;        // 2304 x 768
  bf16* woutT = wqkvT + (size_t)(3 * DIM) * DIM;   //  768 x 768
  bf16* qb    = woutT + (size_t)DIM * DIM;
  bf16* kb    = qb    + (size_t)ROWS * DIM;
  bf16* vb    = kb    + (size_t)ROWS * DIM;
  bf16* attnb = vb    + (size_t)ROWS * DIM;

  // 1) LayerNorm -> bf16
  ln_kernel<<<ROWS, 256, 0, stream>>>(x, ln_scale, ln_bias, xn);

  // 2) Weights: transpose + convert to bf16 (B stored N-major, K-contiguous)
  transpose_to_bf16<<<dim3((3 * DIM) / 32, DIM / 32), 256, 0, stream>>>(
      w_qkv, wqkvT, DIM, 3 * DIM);
  transpose_to_bf16<<<dim3(DIM / 32, DIM / 32), 256, 0, stream>>>(
      w_out, woutT, DIM, DIM);

  // 3) QKV GEMM: (16384x768) x (768x2304) -> deinterleaved q/k/v bf16
  gemm_bf16_kernel<0><<<dim3((3 * DIM) / 128, ROWS / 128), 256, 0, stream>>>(
      xn, wqkvT, 3 * DIM, nullptr, nullptr, qb, kb, vb);

  // 4) Windowed attention
  attn_kernel<<<dim3(SEQ / 128, BATCH * HEADS), 128, 0, stream>>>(
      qb, kb, vb, log_temp, attnb);

  // 5) Output projection: (16384x768) x (768x768) + bias -> f32 out
  gemm_bf16_kernel<1><<<dim3(DIM / 128, ROWS / 128), 256, 0, stream>>>(
      attnb, woutT, DIM, out, b_out, nullptr, nullptr, nullptr);
}
